// GPTJAttention_47897475285586
// MI455X (gfx1250) — matrix operand order repaired and checked by reading, not compiled, not run on hardware
//
#include <hip/hip_runtime.h>
#include <hip/hip_bf16.h>

// ---------------------------------------------------------------------------
// GPT-J attention for MI455X (gfx1250, wave32, WMMA bf16 16x16x32).
//   B=2, S=2048, D=4096, H=16, HD=256, RD=64 (GPT-J interleaved RoPE).
// Pipeline: fp32->bf16 once (w_qkv bf16 = 100MB -> L2-resident), WMMA QKV
// GEMM, RoPE, WMMA flash attention, WMMA out-proj.
// Vectorized b128 LDS fragment loads (transposed staging for contraction-
// major operands), async global->LDS staging (ASYNCcnt) via the gfx1250
// builtin (typed int4 pointers per the toolchain's signature), and
// global_prefetch of next tiles.
// ---------------------------------------------------------------------------

typedef __attribute__((ext_vector_type(16))) __bf16 v16bf;
typedef __attribute__((ext_vector_type(8)))  __bf16 v8bf;
typedef __attribute__((ext_vector_type(8)))  float  v8f;
typedef int v4i __attribute__((vector_size(16)));

namespace cfg {
constexpr int B  = 2;
constexpr int S  = 2048;
constexpr int D  = 4096;
constexpr int H  = 16;
constexpr int HD = 256;
constexpr int RD = 64;
constexpr int BS = B * S;          // 4096 rows
constexpr int N3 = 3 * D;          // 12288
}

// ------------------- async global->LDS staging helpers ---------------------

#if __has_builtin(__builtin_amdgcn_global_load_async_to_lds_b128)
#define ATH_ASYNC_LDS 1
#endif

// Copy 16 bytes (one lane) global -> LDS.
__device__ inline void stage16(const __bf16* g, __bf16* l) {
#ifdef ATH_ASYNC_LDS
  __builtin_amdgcn_global_load_async_to_lds_b128(
      (__attribute__((address_space(1))) v4i*)(uintptr_t)g,
      (__attribute__((address_space(3))) v4i*)(uint32_t)(uintptr_t)l,
      0, 0);
#else
  *(v8bf*)l = *(const v8bf*)g;
#endif
}

__device__ inline void stage_wait() {
#ifdef ATH_ASYNC_LDS
#if __has_builtin(__builtin_amdgcn_s_wait_asynccnt)
  __builtin_amdgcn_s_wait_asynccnt(0);
#else
  asm volatile("s_wait_asynccnt 0" ::: "memory");
#endif
#endif
}

// ---------------- WMMA fragment loaders (ISA 7.12.2 layouts) ---------------
// All LDS row strides are multiples of 8 elements (16B) so every fragment
// read is a pair of ds_load_b128.

// A fragment: 16x32 bf16 (rows M, contraction K), row-major storage.
__device__ inline v16bf frag_a(const __bf16* base, int ld) {
  int lane = threadIdx.x & 31;
  const __bf16* p = base + (lane & 15) * ld + ((lane >> 4) << 3);
  v8bf lo = *(const v8bf*)p;
  v8bf hi = *(const v8bf*)(p + 16);
  return __builtin_shufflevector(lo, hi, 0,1,2,3,4,5,6,7,8,9,10,11,12,13,14,15);
}

// B fragment: 32x16 bf16 (contraction K, cols N) from K-major (transposed)
// storage: memory = Bt[n][k].  lane 0-15: N=lane, K 0..7 and 16..23;
// lane 16-31: N=lane-16, K 8..15 and 24..31.
__device__ inline v16bf frag_bt(const __bf16* base, int ld) {
  int lane = threadIdx.x & 31;
  const __bf16* p = base + (lane & 15) * ld + ((lane >> 4) << 3);
  v8bf lo = *(const v8bf*)p;
  v8bf hi = *(const v8bf*)(p + 16);
  return __builtin_shufflevector(lo, hi, 0,1,2,3,4,5,6,7,8,9,10,11,12,13,14,15);
}

// ---------------------------- fp32 -> bf16 ---------------------------------

__global__ __launch_bounds__(256) void cvt_bf16_kernel(const float* __restrict__ in,
                                                       __bf16* __restrict__ out,
                                                       size_t n) {
  size_t i = ((size_t)blockIdx.x * blockDim.x + threadIdx.x) * 4;
  if (i + 3 < n) {
#pragma unroll
    for (int j = 0; j < 4; ++j) out[i + j] = (__bf16)in[i + j];
  }
}

// ------------------------------ WMMA GEMM ----------------------------------
// C[M,N] = A[M,K] @ B[K,N]; block tile 64x128, 8 waves (2x4), wave 32x32,
// K-step 32.  A staged row-major (async b128), B staged transposed so
// fragment reads are contiguous b128.
// mode 0: scatter bf16 into q/k/v head-major [B,H,S,HD];  mode 1: fp32 out.

__global__ __launch_bounds__(256) void gemm_wmma_kernel(
    const __bf16* __restrict__ A, const __bf16* __restrict__ Bw,
    int M, int N, int K, int mode,
    float* __restrict__ outf,
    __bf16* __restrict__ qb, __bf16* __restrict__ kb, __bf16* __restrict__ vb) {
  using namespace cfg;
  __shared__ __align__(16) __bf16 sA[64][40];     // 64x32 row-major (+pad)
  __shared__ __align__(16) __bf16 sBt[128][40];   // 32x128 stored as [n][k]

  const int tid  = threadIdx.x;
  const int wave = tid >> 5;
  const int wm   = wave >> 2;          // 0..1
  const int wn   = wave & 3;           // 0..3
  const int m0   = blockIdx.y * 64;
  const int n0   = blockIdx.x * 128;

  v8f acc[2][2] = {};

  for (int k0 = 0; k0 < K; k0 += 32) {
    {   // A tile 64x32: one 16B async chunk per thread
      int r = tid >> 2;
      int c = (tid & 3) << 3;
      stage16(A + (size_t)(m0 + r) * K + k0 + c, &sA[r][c]);
    }
    {   // B tile 32x128: vector global load, scalar transpose into sBt
#pragma unroll
      for (int t = 0; t < 2; ++t) {
        int c2 = tid * 2 + t;
        int r  = c2 >> 4;              // k row 0..31
        int c  = (c2 & 15) << 3;       // n col chunk
        v8bf val = *(const v8bf*)(Bw + (size_t)(k0 + r) * N + n0 + c);
#pragma unroll
        for (int i = 0; i < 8; ++i) sBt[c + i][r] = val[i];
      }
    }
    stage_wait();
    __syncthreads();

    v16bf af[2], bf[2];
#pragma unroll
    for (int i = 0; i < 2; ++i) af[i] = frag_a(&sA[wm * 32 + i * 16][0], 40);
#pragma unroll
    for (int j = 0; j < 2; ++j) bf[j] = frag_bt(&sBt[wn * 32 + j * 16][0], 40);
#pragma unroll
    for (int i = 0; i < 2; ++i)
#pragma unroll
      for (int j = 0; j < 2; ++j)
        acc[i][j] = __builtin_amdgcn_wmma_f32_16x16x32_bf16(
            false, af[i], false, bf[j], (short)0, acc[i][j], false, false);
    __syncthreads();
  }

  // epilogue: C layout -> lane holds col nn, rows mb..mb+7 in v8f slots
  const int lane = tid & 31;
  const int nn = lane & 15;
  const int mb = (lane >> 4) << 3;
#pragma unroll
  for (int i = 0; i < 2; ++i) {
#pragma unroll
    for (int j = 0; j < 2; ++j) {
      int gmb = m0 + wm * 32 + i * 16 + mb;
      int gc  = n0 + wn * 32 + j * 16 + nn;
#pragma unroll
      for (int v = 0; v < 8; ++v) {
        int gm = gmb + v;
        float val = acc[i][j][v];
        if (mode == 0) {
          int which = gc >> 12;          // /D : 0=q 1=k 2=v
          int rem   = gc & (D - 1);
          int h     = rem >> 8;          // /HD
          int d     = rem & (HD - 1);
          int bb    = gm >> 11;          // /S
          int ss    = gm & (S - 1);
          size_t idx = ((size_t)(bb * H + h) * S + ss) * HD + d;
          __bf16 bv = (__bf16)val;
          if (which == 0)      qb[idx] = bv;
          else if (which == 1) kb[idx] = bv;
          else                 vb[idx] = bv;
        } else {
          outf[(size_t)gm * N + gc] = val;
        }
      }
    }
  }
}

// ------------------------------- RoPE --------------------------------------

__global__ __launch_bounds__(256) void rope_kernel(__bf16* __restrict__ qb,
                                                   __bf16* __restrict__ kb,
                                                   const int* __restrict__ pos) {
  using namespace cfg;
  int idx = blockIdx.x * blockDim.x + threadIdx.x;   // B*H*S*(RD/2) threads
  int i  = idx & 31;                 // pair index, RD/2 = 32
  int ss = (idx >> 5) & (S - 1);
  int hh = (idx >> 16) & (H - 1);
  int bb = idx >> 20;
  if (bb >= B) return;

  float p = (float)pos[bb * S + ss];
  float inv = __expf(-((float)(2 * i) / (float)RD) * 9.210340371976184f);
  float ang = p * inv;
  float c, sn;
  __sincosf(ang, &sn, &c);

  size_t base = ((size_t)(bb * H + hh) * S + ss) * HD + 2 * i;
  {
    float x0 = (float)qb[base], x1 = (float)qb[base + 1];
    qb[base]     = (__bf16)(x0 * c - x1 * sn);
    qb[base + 1] = (__bf16)(x1 * c + x0 * sn);
  }
  {
    float x0 = (float)kb[base], x1 = (float)kb[base + 1];
    kb[base]     = (__bf16)(x0 * c - x1 * sn);
    kb[base + 1] = (__bf16)(x1 * c + x0 * sn);
  }
}

// --------------------------- Flash attention -------------------------------
// grid = (S/64, B*H), block = 128 (4 waves). Wave w owns q rows [q0+16w, +16).
// Q/K staged row-major (async b128; contraction d is contiguous), V staged
// transposed [d][k] so PV fragments are contiguous b128.

__global__ __launch_bounds__(128) void flash_attn_kernel(
    const __bf16* __restrict__ qg, const __bf16* __restrict__ kg,
    const __bf16* __restrict__ vg, __bf16* __restrict__ ctx) {
  using namespace cfg;
  __shared__ __align__(16) __bf16 sQ[64][HD + 8];
  __shared__ __align__(16) __bf16 sK[32][HD + 8];
  __shared__ __align__(16) __bf16 sVt[HD][40];      // V transposed [d][k]
  __shared__ __align__(16) __bf16 sP[4][16][40];

  const int tid  = threadIdx.x;
  const int wave = tid >> 5;
  const int lane = tid & 31;
  const int nn = lane & 15;
  const int mb = (lane >> 4) << 3;

  const int q0 = blockIdx.x * 64;
  const int bh = blockIdx.y;           // b*H + h
  const int bb = bh >> 4;
  const int hh = bh & (H - 1);

  const __bf16* Qg = qg + (size_t)bh * S * HD;
  const __bf16* Kg = kg + (size_t)bh * S * HD;
  const __bf16* Vg = vg + (size_t)bh * S * HD;

  // stage Q tile 64x256 (16 async 16B chunks per thread)
#pragma unroll
  for (int t = 0; t < 16; ++t) {
    int c2 = tid * 16 + t;
    int r  = c2 >> 5;
    int c  = (c2 & 31) << 3;
    stage16(Qg + (size_t)(q0 + r) * HD + c, &sQ[r][c]);
  }

  v8f acc[16] = {};
  float mrow[8], lrow[8];
#pragma unroll
  for (int v = 0; v < 8; ++v) { mrow[v] = -1e30f; lrow[v] = 0.0f; }

  const int qrow0 = q0 + wave * 16;
  const int nk = q0 + 64;              // causal: keys 0 .. q0+63
  const float sc = 0.0625f;            // 1/sqrt(HD)

  for (int k0 = 0; k0 < nk; k0 += 32) {
    __syncthreads();
    // prefetch next K/V tile rows while staging this one
    if (k0 + 32 < nk) {
      __builtin_prefetch(Kg + (size_t)(k0 + 32 + (tid & 31)) * HD, 0, 1);
      __builtin_prefetch(Vg + (size_t)(k0 + 32 + (tid & 31)) * HD, 0, 1);
    }
    // stage K tile 32x256 row-major (8 async chunks/thread) and
    // V tile 32x256 transposed into sVt (vector load + scalar scatter)
#pragma unroll
    for (int t = 0; t < 8; ++t) {
      int c2 = tid * 8 + t;
      int r  = c2 >> 5;
      int c  = (c2 & 31) << 3;
      stage16(Kg + (size_t)(k0 + r) * HD + c, &sK[r][c]);
      v8bf val = *(const v8bf*)(Vg + (size_t)(k0 + r) * HD + c);
#pragma unroll
      for (int i = 0; i < 8; ++i) sVt[c + i][r] = val[i];
    }
    stage_wait();
    __syncthreads();

    // scores: S[16 x 32] = Q[16 x 256] @ K^T, two 16-col n-tiles
    v8f s0 = {}, s1 = {};
#pragma unroll
    for (int ks = 0; ks < 8; ++ks) {
      v16bf qa = frag_a(&sQ[wave * 16][ks * 32], HD + 8);
      v16bf b0 = frag_bt(&sK[0][ks * 32], HD + 8);
      v16bf b1 = frag_bt(&sK[16][ks * 32], HD + 8);
      s0 = __builtin_amdgcn_wmma_f32_16x16x32_bf16(false, qa, false, b0, (short)0, s0, false, false);
      s1 = __builtin_amdgcn_wmma_f32_16x16x32_bf16(false, qa, false, b1, (short)0, s1, false, false);
    }

    // online softmax update (rows 0-7 in lanes 0-15, rows 8-15 in 16-31;
    // xor-reduce over 8..1 stays within each half)
    float p0[8], p1[8], alpha[8];
#pragma unroll
    for (int v = 0; v < 8; ++v) {
      float x0 = s0[v] * sc, x1 = s1[v] * sc;
      int qr = qrow0 + mb + v;
      if (k0 + nn > qr)      x0 = -1e30f;
      if (k0 + 16 + nn > qr) x1 = -1e30f;
      float t = fmaxf(x0, x1);
#pragma unroll
      for (int off = 8; off; off >>= 1) t = fmaxf(t, __shfl_xor(t, off, 32));
      float mnew = fmaxf(mrow[v], t);
      float a  = __expf(mrow[v] - mnew);
      float e0 = __expf(x0 - mnew);
      float e1 = __expf(x1 - mnew);
      float rs = e0 + e1;
#pragma unroll
      for (int off = 8; off; off >>= 1) rs += __shfl_xor(rs, off, 32);
      lrow[v] = lrow[v] * a + rs;
      mrow[v] = mnew;
      alpha[v] = a;
      p0[v] = e0; p1[v] = e1;
    }
#pragma unroll
    for (int t = 0; t < 16; ++t)
#pragma unroll
      for (int v = 0; v < 8; ++v) acc[t][v] *= alpha[v];

    // transpose P (C layout -> A layout) through per-wave LDS pad
#pragma unroll
    for (int v = 0; v < 8; ++v) {
      sP[wave][mb + v][nn]      = (__bf16)p0[v];
      sP[wave][mb + v][16 + nn] = (__bf16)p1[v];
    }
    v16bf pf = frag_a(&sP[wave][0][0], 40);

    // ctx[16 x 256] += P[16 x 32] @ V[32 x 256]
#pragma unroll
    for (int t = 0; t < 16; ++t) {
      v16bf bv = frag_bt(&sVt[t * 16][0], 40);
      acc[t] = __builtin_amdgcn_wmma_f32_16x16x32_bf16(false, pf, false, bv, (short)0, acc[t], false, false);
    }
  }

  // normalize and store ctx as bf16 [B*S, D] (col = h*HD + d)
  float linv[8];
#pragma unroll
  for (int v = 0; v < 8; ++v) linv[v] = 1.0f / lrow[v];
#pragma unroll
  for (int t = 0; t < 16; ++t) {
#pragma unroll
    for (int v = 0; v < 8; ++v) {
      int qr = qrow0 + mb + v;
      size_t idx = (size_t)(bb * S + qr) * D + hh * HD + t * 16 + nn;
      ctx[idx] = (__bf16)(acc[t][v] * linv[v]);
    }
  }
}

// ------------------------------- launch ------------------------------------

extern "C" void kernel_launch(void* const* d_in, const int* in_sizes, int n_in,
                              void* d_out, int out_size, void* d_ws, size_t ws_size,
                              hipStream_t stream) {
  using namespace cfg;
  (void)in_sizes; (void)n_in; (void)out_size; (void)ws_size;

  const int*   pos    = (const int*)d_in[0];
  const float* hidden = (const float*)d_in[1];
  const float* wqkv   = (const float*)d_in[2];
  const float* wout   = (const float*)d_in[3];
  float*       out    = (float*)d_out;

  char* ws = (char*)d_ws;
  size_t off = 0;
  __bf16* hb   = (__bf16*)(ws + off); off += (size_t)BS * D * 2;   // hidden bf16
  __bf16* wqb  = (__bf16*)(ws + off); off += (size_t)D * N3 * 2;   // w_qkv bf16
  __bf16* wob  = (__bf16*)(ws + off); off += (size_t)D * D * 2;    // w_out bf16
  __bf16* qb   = (__bf16*)(ws + off); off += (size_t)BS * D * 2;   // Q [B,H,S,HD]
  __bf16* kb   = (__bf16*)(ws + off); off += (size_t)BS * D * 2;   // K
  __bf16* vb   = (__bf16*)(ws + off); off += (size_t)BS * D * 2;   // V
  __bf16* ctxb = (__bf16*)(ws + off); off += (size_t)BS * D * 2;   // ctx bf16

  {
    size_t n1 = (size_t)BS * D;     // 16M
    size_t n2 = (size_t)D * N3;     // 50M
    size_t n3 = (size_t)D * D;      // 16M
    cvt_bf16_kernel<<<dim3((unsigned)(n1 / 4 / 256)), 256, 0, stream>>>(hidden, hb, n1);
    cvt_bf16_kernel<<<dim3((unsigned)(n2 / 4 / 256)), 256, 0, stream>>>(wqkv, wqb, n2);
    cvt_bf16_kernel<<<dim3((unsigned)(n3 / 4 / 256)), 256, 0, stream>>>(wout, wob, n3);
  }

  gemm_wmma_kernel<<<dim3(N3 / 128, BS / 64), 256, 0, stream>>>(
      hb, wqb, BS, N3, D, 0, nullptr, qb, kb, vb);

  {
    int total = B * H * S * (RD / 2);   // 2,097,152
    rope_kernel<<<total / 256, 256, 0, stream>>>(qb, kb, pos);
  }

  flash_attn_kernel<<<dim3(S / 64, B * H), 128, 0, stream>>>(qb, kb, vb, ctxb);

  gemm_wmma_kernel<<<dim3(D / 128, BS / 64), 256, 0, stream>>>(
      ctxb, wob, BS, D, D, 1, out, nullptr, nullptr, nullptr);
}
